// MotionMALA_69715909148877
// MI455X (gfx1250) — matrix operand order repaired and checked by reading, not compiled, not run on hardware
//
#include <hip/hip_runtime.h>
#include <hip/hip_bf16.h>
#include <math.h>

// ---------------------------------------------------------------------------
// MI455X (gfx1250) edge-linear-attention transformer layer.
// Bandwidth-bound: linear-attention reformulation avoids the [E,H,D,D]
// outer-product scatter (1 GB -> ~70 MB of L2-resident f32 atomics).
// All GEMMs: v_wmma_f32_16x16x32_bf16 with pre-packed bf16 fragments:
//   - weights packed once per launch into B-fragment order; fragments are
//     preloaded 4-k-blocks-ahead so b128 loads overlap the WMMA chain
//   - activations written into A-fragment order in LDS (32B/lane ds loads)
// Per-matrix tile loops + uniform full-tile paths keep EXEC churn out of
// the hot epilogues.
// ---------------------------------------------------------------------------

typedef __attribute__((ext_vector_type(16))) __bf16 v16bf;
typedef __attribute__((ext_vector_type(8)))  float  v8f;

#define HIDDIM 128
#define NHEAD  8
#define HD     16

__device__ __forceinline__ float phi_f(float x)     { return x > 0.f ? x + 1.f : __expf(x); }
__device__ __forceinline__ float sigmoid_f(float x) { return 1.f / (1.f + __expf(-x)); }
__device__ __forceinline__ float gelu_f(float x)    { return 0.5f * x * (1.f + erff(x * 0.70710678118654752f)); }

// 8 consecutive lanes (sub = tid&7) cooperate on one 128-wide row.
__device__ __forceinline__ void row_stats(const float* vals, float& mu, float& rs) {
  float s = 0.f, s2 = 0.f;
#pragma unroll
  for (int i = 0; i < 16; ++i) { s += vals[i]; s2 += vals[i] * vals[i]; }
#pragma unroll
  for (int m = 1; m < 8; m <<= 1) { s += __shfl_xor(s, m, 8); s2 += __shfl_xor(s2, m, 8); }
  mu = s * (1.f / 128.f);
  float var = s2 * (1.f / 128.f) - mu * mu;
  rs = rsqrtf(var + 1e-5f);
}

// Scatter one A element (row 0..15, K index c) into packed A-fragment layout
// (CDNA5 ISA 7.12.2, 16-bit A 16x32): Apk[(kb*32 + lane)*16 + elem].
__device__ __forceinline__ void stA(__bf16* Apk, int row, int c, __bf16 v) {
  const int kb = c >> 5, kk = c & 31, rem = kk & 15;
  const int lane = row + ((rem >> 3) << 4);
  const int elem = ((kk >> 4) << 3) + (rem & 7);
  Apk[(((kb << 5) + lane) << 4) + elem] = v;
}

// One wave computes a 16x16 fp32 tile from packed fragments.
// Processes K in chunks of 4 k-blocks (K=128): preload all 8 fragments of the
// chunk (4 A from LDS, 4 B from global) before the 4 chained WMMAs so the
// b128 loads issue as one clause and overlap the matrix pipe.
template <int KB>
__device__ __forceinline__ v8f wmma_acc_pk(const __bf16* Apk, const __bf16* Bpk, v8f c) {
  static_assert(KB % 4 == 0, "KB must be a multiple of 4");
  const int lane = threadIdx.x & 31;
#pragma unroll
  for (int kc = 0; kc < KB; kc += 4) {
    v16bf a0 = *(const v16bf*)(Apk + (((kc + 0) * 32 + lane) << 4));
    v16bf a1 = *(const v16bf*)(Apk + (((kc + 1) * 32 + lane) << 4));
    v16bf a2 = *(const v16bf*)(Apk + (((kc + 2) * 32 + lane) << 4));
    v16bf a3 = *(const v16bf*)(Apk + (((kc + 3) * 32 + lane) << 4));
    v16bf b0 = *(const v16bf*)(Bpk + (((kc + 0) * 32 + lane) << 4));
    v16bf b1 = *(const v16bf*)(Bpk + (((kc + 1) * 32 + lane) << 4));
    v16bf b2 = *(const v16bf*)(Bpk + (((kc + 2) * 32 + lane) << 4));
    v16bf b3 = *(const v16bf*)(Bpk + (((kc + 3) * 32 + lane) << 4));
    c = __builtin_amdgcn_wmma_f32_16x16x32_bf16(false, a0, false, b0, (short)0, c, false, false);
    c = __builtin_amdgcn_wmma_f32_16x16x32_bf16(false, a1, false, b1, (short)0, c, false, false);
    c = __builtin_amdgcn_wmma_f32_16x16x32_bf16(false, a2, false, b2, (short)0, c, false, false);
    c = __builtin_amdgcn_wmma_f32_16x16x32_bf16(false, a3, false, b3, (short)0, c, false, false);
  }
  return c;
}

__global__ void k_zero(float* __restrict__ p, long n) {
  long i = (long)blockIdx.x * blockDim.x + threadIdx.x;
  if (i < n) p[i] = 0.f;
}

// Pack fp32 weight [K x ncols] into per-tile B-fragment bf16 layout.
__global__ void k_pack_w(const float* __restrict__ W, __bf16* __restrict__ P,
                         int K, int ncols) {
  int idx = blockIdx.x * blockDim.x + threadIdx.x;
  if (idx >= K * ncols) return;
  const int i    = idx & 15;
  const int lane = (idx >> 4) & 31;
  const int rest = idx >> 9;
  const int kblocks = K >> 5;
  const int kb = rest % kblocks;
  const int t  = rest / kblocks;
  const int k   = (kb << 5) + ((i >> 3) << 4) + ((lane >> 4) << 3) + (i & 7);
  const int col = (t << 4) + (lane & 15);
  P[idx] = (__bf16)W[(size_t)k * ncols + col];
}

// --- Kernel 1: xs = LN(x); phi_q | k | v ------------------------------------
__global__ void k_node_qkv(const float* __restrict__ x,
                           const float* __restrict__ lng, const float* __restrict__ lnb,
                           const __bf16* __restrict__ pQ, const __bf16* __restrict__ pK,
                           const __bf16* __restrict__ pV,
                           float* __restrict__ xs, float* __restrict__ phiq,
                           float* __restrict__ kbuf, float* __restrict__ vbuf, int N) {
  __shared__ __attribute__((aligned(32))) __bf16 Apk[4 * 32 * 16];
  const int tid = threadIdx.x;
  const int node0 = blockIdx.x * 16;
  const bool full = (node0 + 16 <= N);          // uniform fast path
  {
    const int rrow = tid >> 3, sub = tid & 7;
    const int nd = full ? (node0 + rrow) : min(node0 + rrow, N - 1);
    float vals[16];
#pragma unroll
    for (int i = 0; i < 16; ++i) vals[i] = x[(size_t)nd * HIDDIM + sub * 16 + i];
    float mu, rs; row_stats(vals, mu, rs);
    const bool wr = full || (node0 + rrow < N);
#pragma unroll
    for (int i = 0; i < 16; ++i) {
      int c = sub * 16 + i;
      float y = (vals[i] - mu) * rs * lng[c] + lnb[c];
      stA(Apk, rrow, c, (__bf16)y);
      if (wr) xs[(size_t)nd * HIDDIM + c] = y;
    }
  }
  __syncthreads();
  const int w = tid >> 5, lane = tid & 31;
  const int nloc = lane & 15, mb = (lane >> 4) << 3;

  // Q tiles (phi applied, straight-line epilogue)
  for (int tt = w; tt < 8; tt += 4) {
    const int col0 = tt << 4;
    v8f c = {};
    c = wmma_acc_pk<4>(Apk, pQ + (size_t)tt * (4 * 512), c);
    float* cp = (float*)&c;
    if (full) {
#pragma unroll
      for (int j = 0; j < 8; ++j)
        phiq[(size_t)(node0 + mb + j) * HIDDIM + col0 + nloc] = phi_f(cp[j]);
    } else {
#pragma unroll
      for (int j = 0; j < 8; ++j)
        if (node0 + mb + j < N)
          phiq[(size_t)(node0 + mb + j) * HIDDIM + col0 + nloc] = phi_f(cp[j]);
    }
  }
  // K tiles
  for (int tt = w; tt < 8; tt += 4) {
    const int col0 = tt << 4;
    v8f c = {};
    c = wmma_acc_pk<4>(Apk, pK + (size_t)tt * (4 * 512), c);
    float* cp = (float*)&c;
    if (full) {
#pragma unroll
      for (int j = 0; j < 8; ++j)
        kbuf[(size_t)(node0 + mb + j) * HIDDIM + col0 + nloc] = cp[j];
    } else {
#pragma unroll
      for (int j = 0; j < 8; ++j)
        if (node0 + mb + j < N)
          kbuf[(size_t)(node0 + mb + j) * HIDDIM + col0 + nloc] = cp[j];
    }
  }
  // V tiles
  for (int tt = w; tt < 8; tt += 4) {
    const int col0 = tt << 4;
    v8f c = {};
    c = wmma_acc_pk<4>(Apk, pV + (size_t)tt * (4 * 512), c);
    float* cp = (float*)&c;
    if (full) {
#pragma unroll
      for (int j = 0; j < 8; ++j)
        vbuf[(size_t)(node0 + mb + j) * HIDDIM + col0 + nloc] = cp[j];
    } else {
#pragma unroll
      for (int j = 0; j < 8; ++j)
        if (node0 + mb + j < N)
          vbuf[(size_t)(node0 + mb + j) * HIDDIM + col0 + nloc] = cp[j];
    }
  }
}

// --- Kernel 2: per-edge GEMM + phi + coeff + scatter ------------------------
__global__ void k_edge_scatter(const float* __restrict__ r, const int* __restrict__ eidx,
                               const float* __restrict__ lng, const float* __restrict__ lnb,
                               const __bf16* __restrict__ pKr, const __bf16* __restrict__ pVr,
                               const float* __restrict__ kbuf, const float* __restrict__ vbuf,
                               const float* __restrict__ phiq,
                               float* __restrict__ num, float* __restrict__ den,
                               int E, int N) {
  __shared__ __attribute__((aligned(32))) __bf16 Apk[4 * 32 * 16];
  __shared__ float ke[16][HIDDIM];
  __shared__ float ve[16][HIDDIM];
  __shared__ int   s_src[16], s_dst[16];
  const int tid = threadIdx.x;
  const int e0 = blockIdx.x * 16;
  const bool full = (e0 + 16 <= E);             // uniform fast path
  if (tid < 16) {
    int e = min(e0 + tid, E - 1);
    s_src[tid] = min(max(eidx[e], 0), N - 1);
    s_dst[tid] = min(max(eidx[E + e], 0), N - 1);
  }
  {
    const int rrow = tid >> 3, sub = tid & 7;
    const int er = full ? (e0 + rrow) : min(e0 + rrow, E - 1);
    float vals[16];
#pragma unroll
    for (int i = 0; i < 16; ++i) vals[i] = r[(size_t)er * HIDDIM + sub * 16 + i];
    float mu, rs; row_stats(vals, mu, rs);
#pragma unroll
    for (int i = 0; i < 16; ++i) {
      int c = sub * 16 + i;
      stA(Apk, rrow, c, (__bf16)((vals[i] - mu) * rs * lng[c] + lnb[c]));
    }
  }
  __syncthreads();
  const int w = tid >> 5, lane = tid & 31;
  const int nloc = lane & 15, mb = (lane >> 4) << 3;

  // kr tiles: k_e = k[src] + rn@Wkr, then phi
  for (int tt = w; tt < 8; tt += 4) {
    const int col0 = tt << 4;
    v8f c = {};
    c = wmma_acc_pk<4>(Apk, pKr + (size_t)tt * (4 * 512), c);
    float* cp = (float*)&c;
#pragma unroll
    for (int j = 0; j < 8; ++j) {
      int m = mb + j;
      ke[m][col0 + nloc] =
          phi_f(cp[j] + kbuf[(size_t)s_src[m] * HIDDIM + col0 + nloc]);
    }
  }
  // vr tiles: v_e = v[src] + rn@Wvr
  for (int tt = w; tt < 8; tt += 4) {
    const int col0 = tt << 4;
    v8f c = {};
    c = wmma_acc_pk<4>(Apk, pVr + (size_t)tt * (4 * 512), c);
    float* cp = (float*)&c;
#pragma unroll
    for (int j = 0; j < 8; ++j) {
      int m = mb + j;
      ve[m][col0 + nloc] = cp[j] + vbuf[(size_t)s_src[m] * HIDDIM + col0 + nloc];
    }
  }
  __syncthreads();
  {
    const int e = tid >> 3, h = tid & 7;               // 16 edges x 8 heads
    if (full || (e0 + e < E)) {
      const int dn = s_dst[e];
      const float* pq = phiq + (size_t)dn * HIDDIM + h * HD;
      float coeff = 0.f;
#pragma unroll
      for (int d = 0; d < HD; ++d) coeff += ke[e][h * HD + d] * pq[d];
      atomicAdd(&den[(size_t)dn * NHEAD + h], coeff);
      float* np = num + (size_t)dn * HIDDIM + h * HD;
#pragma unroll
      for (int d = 0; d < HD; ++d) atomicAdd(&np[d], coeff * ve[e][h * HD + d]);
    }
  }
}

// --- Kernel 3: out = (num/den)@Wo; gate; x_mid = x + LN(attn); xmln ---------
__global__ void k_node_out(const float* __restrict__ x,  const float* __restrict__ xs,
                           const float* __restrict__ num, const float* __restrict__ den,
                           const __bf16* __restrict__ pWo,
                           const __bf16* __restrict__ pWg, const float* __restrict__ bg,
                           const __bf16* __restrict__ pWs, const float* __restrict__ bs,
                           const float* __restrict__ lnpg, const float* __restrict__ lnpb,
                           const float* __restrict__ lnfg, const float* __restrict__ lnfb,
                           float* __restrict__ xmid, __bf16* __restrict__ xmln, int N) {
  __shared__ __attribute__((aligned(32))) __bf16 a_op [4 * 32 * 16];
  __shared__ __attribute__((aligned(32))) __bf16 a_cat[8 * 32 * 16]; // [out | xs]
  __shared__ float outF[16][HIDDIM];
  __shared__ float attnF[16][HIDDIM];
  const int tid = threadIdx.x;
  const int node0 = blockIdx.x * 16;
  const bool full = (node0 + 16 <= N);
  const int rrow = tid >> 3, sub = tid & 7;
  const int nd = full ? (node0 + rrow) : min(node0 + rrow, N - 1);
#pragma unroll
  for (int i = 0; i < 16; ++i) {
    int c = sub * 16 + i;
    float dv = fmaxf(den[(size_t)nd * NHEAD + (c >> 4)], 1e-6f);
    stA(a_op, rrow, c, (__bf16)(num[(size_t)nd * HIDDIM + c] / dv));
    stA(a_cat, rrow, 128 + c, (__bf16)xs[(size_t)nd * HIDDIM + c]);  // xs half
  }
  __syncthreads();
  const int w = tid >> 5, lane = tid & 31;
  const int nloc = lane & 15, mb = (lane >> 4) << 3;
  for (int tt = w; tt < 8; tt += 4) {                  // out = op @ Wo
    const int col0 = tt << 4;
    v8f c = {};
    c = wmma_acc_pk<4>(a_op, pWo + (size_t)tt * (4 * 512), c);
    float* cp = (float*)&c;
#pragma unroll
    for (int j = 0; j < 8; ++j) {
      int m = mb + j, cc = col0 + nloc;
      outF[m][cc] = cp[j];
      stA(a_cat, m, cc, (__bf16)cp[j]);                // out half of concat
    }
  }
  __syncthreads();
  for (int tt = w; tt < 8; tt += 4) {                  // gate (K=256) + skip
    const int col0 = tt << 4;
    v8f gc = {}, sc = {};
    gc = wmma_acc_pk<8>(a_cat,           pWg + (size_t)tt * (8 * 512), gc);
    sc = wmma_acc_pk<4>(a_cat + 4 * 512, pWs + (size_t)tt * (4 * 512), sc);
    float* gp = (float*)&gc; float* sp = (float*)&sc;
#pragma unroll
    for (int j = 0; j < 8; ++j) {
      int m = mb + j, cc = col0 + nloc;
      float o = outF[m][cc];
      float g = sigmoid_f(gp[j] + bg[cc]);
      float s = sp[j] + bs[cc];
      attnF[m][cc] = o + g * (s - o);
    }
  }
  __syncthreads();
  {
    float vals[16];
#pragma unroll
    for (int i = 0; i < 16; ++i) vals[i] = attnF[rrow][sub * 16 + i];
    float mu, rs; row_stats(vals, mu, rs);
    float xm[16];
#pragma unroll
    for (int i = 0; i < 16; ++i) {
      int c = sub * 16 + i;
      xm[i] = x[(size_t)nd * HIDDIM + c] + (vals[i] - mu) * rs * lnpg[c] + lnpb[c];
    }
    float mu2, rs2; row_stats(xm, mu2, rs2);
    const bool wr = full || (node0 + rrow < N);
#pragma unroll
    for (int i = 0; i < 16; ++i) {
      int c = sub * 16 + i;
      if (wr) {
        xmid[(size_t)nd * HIDDIM + c] = xm[i];
        xmln[(size_t)nd * HIDDIM + c] = (__bf16)((xm[i] - mu2) * rs2 * lnfg[c] + lnfb[c]);
      }
    }
  }
}

// --- Kernel 4: h1 = gelu(xmln @ W1 + b1) ------------------------------------
__global__ void k_ffn1(const __bf16* __restrict__ xmln,
                       const __bf16* __restrict__ pW1, const float* __restrict__ b1,
                       __bf16* __restrict__ h1, int N) {
  __shared__ __attribute__((aligned(32))) __bf16 Apk[4 * 32 * 16];
  const int tid = threadIdx.x;
  const int node0 = blockIdx.x * 16;
  const bool full = (node0 + 16 <= N);
  {
    const int rrow = tid >> 3, sub = tid & 7;
    const int nd = full ? (node0 + rrow) : min(node0 + rrow, N - 1);
#pragma unroll
    for (int i = 0; i < 16; ++i) {
      int c = sub * 16 + i;
      stA(Apk, rrow, c, xmln[(size_t)nd * HIDDIM + c]);
    }
  }
  __syncthreads();
  const int w = tid >> 5, lane = tid & 31;
  const int nloc = lane & 15, mb = (lane >> 4) << 3;
  for (int tt = w; tt < 32; tt += 4) {                 // 512 output cols
    const int col0 = tt << 4;
    v8f c = {};
    c = wmma_acc_pk<4>(Apk, pW1 + (size_t)tt * (4 * 512), c);
    float* cp = (float*)&c;
    if (full) {
#pragma unroll
      for (int j = 0; j < 8; ++j)
        h1[(size_t)(node0 + mb + j) * (4 * HIDDIM) + col0 + nloc] =
            (__bf16)gelu_f(cp[j] + b1[col0 + nloc]);
    } else {
#pragma unroll
      for (int j = 0; j < 8; ++j) {
        int m = node0 + mb + j;
        if (m < N)
          h1[(size_t)m * (4 * HIDDIM) + col0 + nloc] =
              (__bf16)gelu_f(cp[j] + b1[col0 + nloc]);
      }
    }
  }
}

// --- Kernel 5: final = x_mid + LN(h1 @ W2 + b2) -----------------------------
__global__ void k_ffn2(const __bf16* __restrict__ h1,
                       const __bf16* __restrict__ pW2, const float* __restrict__ b2,
                       const float* __restrict__ lnfg, const float* __restrict__ lnfb,
                       const float* __restrict__ xmid,
                       float* __restrict__ out, int N) {
  __shared__ __attribute__((aligned(32))) __bf16 Apk[16 * 32 * 16]; // K=512
  __shared__ float h2F[16][HIDDIM];
  const int tid = threadIdx.x;
  const int node0 = blockIdx.x * 16;
  const bool full = (node0 + 16 <= N);
  const int rrow = tid >> 3, sub = tid & 7;
  const int nd = full ? (node0 + rrow) : min(node0 + rrow, N - 1);
#pragma unroll
  for (int i = 0; i < 64; ++i) {
    int c = sub * 64 + i;
    stA(Apk, rrow, c, h1[(size_t)nd * (4 * HIDDIM) + c]);
  }
  __syncthreads();
  const int w = tid >> 5, lane = tid & 31;
  const int nloc = lane & 15, mb = (lane >> 4) << 3;
  for (int tt = w; tt < 8; tt += 4) {
    const int col0 = tt << 4;
    v8f c = {};
    c = wmma_acc_pk<16>(Apk, pW2 + (size_t)tt * (16 * 512), c);
    float* cp = (float*)&c;
#pragma unroll
    for (int j = 0; j < 8; ++j)
      h2F[mb + j][col0 + nloc] = cp[j] + b2[col0 + nloc];
  }
  __syncthreads();
  {
    float vals[16];
#pragma unroll
    for (int i = 0; i < 16; ++i) vals[i] = h2F[rrow][sub * 16 + i];
    float mu, rs; row_stats(vals, mu, rs);
    const bool wr = full || (node0 + rrow < N);
#pragma unroll
    for (int i = 0; i < 16; ++i) {
      int c = sub * 16 + i;
      if (wr)
        out[(size_t)nd * HIDDIM + c] =
            xmid[(size_t)nd * HIDDIM + c] + (vals[i] - mu) * rs * lnfg[c] + lnfb[c];
    }
  }
}

extern "C" void kernel_launch(void* const* d_in, const int* in_sizes, int n_in,
                              void* d_out, int out_size, void* d_ws, size_t ws_size,
                              hipStream_t stream) {
  const float* x          = (const float*)d_in[0];
  const float* r          = (const float*)d_in[1];
  const int*   eidx       = (const int*)d_in[2];   // harness ABI: integer -> int32
  const float* ln_x_g     = (const float*)d_in[3];
  const float* ln_x_b     = (const float*)d_in[4];
  const float* ln_r_g     = (const float*)d_in[5];
  const float* ln_r_b     = (const float*)d_in[6];
  const float* ln_post_g  = (const float*)d_in[7];
  const float* ln_post_b  = (const float*)d_in[8];
  const float* ln_ffpre_g = (const float*)d_in[9];
  const float* ln_ffpre_b = (const float*)d_in[10];
  const float* ln_ffpost_g= (const float*)d_in[11];
  const float* ln_ffpost_b= (const float*)d_in[12];
  const float* Wq = (const float*)d_in[13];
  const float* Wk = (const float*)d_in[14];
  const float* Wv = (const float*)d_in[15];
  const float* Wkr= (const float*)d_in[16];
  const float* Wvr= (const float*)d_in[17];
  const float* Wo = (const float*)d_in[18];
  const float* Wg = (const float*)d_in[19];
  const float* bg = (const float*)d_in[20];
  const float* Ws = (const float*)d_in[21];
  const float* bs = (const float*)d_in[22];
  const float* W1 = (const float*)d_in[23];
  const float* b1 = (const float*)d_in[24];
  const float* W2 = (const float*)d_in[25];
  const float* b2 = (const float*)d_in[26];

  const int N = in_sizes[0] / HIDDIM;
  const int E = in_sizes[1] / HIDDIM;

  // ---- workspace layout ----
  float* xs    = (float*)d_ws;
  float* phiq  = xs   + (size_t)N * HIDDIM;
  float* kbuf  = phiq + (size_t)N * HIDDIM;
  float* vbuf  = kbuf + (size_t)N * HIDDIM;
  float* num   = vbuf + (size_t)N * HIDDIM;
  float* den   = num  + (size_t)N * HIDDIM;
  float* xmid  = den  + (size_t)N * NHEAD;
  // packed bf16 weights (B-fragment order); 32B-aligned for all N
  __bf16* pWq  = (__bf16*)(xmid + (size_t)N * HIDDIM);
  __bf16* pWk  = pWq  + 16384;
  __bf16* pWv  = pWk  + 16384;
  __bf16* pWkr = pWv  + 16384;
  __bf16* pWvr = pWkr + 16384;
  __bf16* pWo  = pWvr + 16384;
  __bf16* pWs  = pWo  + 16384;
  __bf16* pWg  = pWs  + 16384;   // K=256 -> 32768
  __bf16* pW1  = pWg  + 32768;   // 128x512 -> 65536
  __bf16* pW2  = pW1  + 65536;   // 512x128 -> 65536
  __bf16* xmln = pW2  + 65536;
  __bf16* h1   = xmln + (size_t)N * HIDDIM;

  // ---- pack weights (cheap, deterministic, every call) ----
  auto pack = [&](const float* W, __bf16* P, int K, int nc) {
    int total = K * nc;
    k_pack_w<<<(total + 255) / 256, 256, 0, stream>>>(W, P, K, nc);
  };
  pack(Wq, pWq, 128, 128);  pack(Wk, pWk, 128, 128);  pack(Wv, pWv, 128, 128);
  pack(Wkr, pWkr, 128, 128); pack(Wvr, pWvr, 128, 128);
  pack(Wo, pWo, 128, 128);  pack(Ws, pWs, 128, 128);
  pack(Wg, pWg, 256, 128);
  pack(W1, pW1, 128, 512);  pack(W2, pW2, 512, 128);

  const long nz = (long)N * (HIDDIM + NHEAD);          // num + den contiguous
  const int nb = (N + 15) / 16, eb = (E + 15) / 16;

  k_zero<<<(int)((nz + 255) / 256), 256, 0, stream>>>(num, nz);
  k_node_qkv<<<nb, 128, 0, stream>>>(x, ln_x_g, ln_x_b, pWq, pWk, pWv,
                                     xs, phiq, kbuf, vbuf, N);
  k_edge_scatter<<<eb, 128, 0, stream>>>(r, eidx, ln_r_g, ln_r_b, pWkr, pWvr,
                                         kbuf, vbuf, phiq, num, den, E, N);
  k_node_out<<<nb, 128, 0, stream>>>(x, xs, num, den, pWo, pWg, bg, pWs, bs,
                                     ln_post_g, ln_post_b, ln_ffpre_g, ln_ffpre_b,
                                     xmid, xmln, N);
  k_ffn1<<<nb, 128, 0, stream>>>(xmln, pW1, b1, h1, N);
  k_ffn2<<<nb, 128, 0, stream>>>(h1, pW2, b2, ln_ffpost_g, ln_ffpost_b, xmid,
                                 (float*)d_out, N);
}
